// SelfAttentionBlock_22376779612588
// MI455X (gfx1250) — matrix operand order repaired and checked by reading, not compiled, not run on hardware
//
#include <hip/hip_runtime.h>
#include <hip/hip_bf16.h>

#define EMBED 1024
#define HEADS 16
#define HDIM  64
#define NB    4
#define SEQL  1024

typedef __attribute__((ext_vector_type(16))) _Float16 v16h;
typedef __attribute__((ext_vector_type(8)))  float    v8f;

static __device__ __forceinline__ v8f wmma_f16(v16h a, v16h b, v8f c) {
  // D = A(16x32 f16) * B(32x16 f16) + C(16x16 f32)
  return __builtin_amdgcn_wmma_f32_16x16x32_f16(false, a, false, b, (short)0, c,
                                                false, false);
}

// ---- fragment loaders (contiguous k = 16*g + h order, wave32) ----
// A 16x32 (MxK), element (m,k) at src[m*ld + k]; lane = 16*g + r holds row r.
// half h -> K = 16*g + h
static __device__ __forceinline__ v16h load_a_f16(const _Float16* src, int ld,
                                                  int g, int r) {
  v16h a;
#pragma unroll
  for (int h = 0; h < 16; ++h) {
    int k = (g << 4) + h;
    a[h] = src[r * ld + k];
  }
  return a;
}
static __device__ __forceinline__ v16h load_a_f32(const float* src, int ld,
                                                  int g, int r) {
  v16h a;
#pragma unroll
  for (int h = 0; h < 16; ++h) {
    int k = (g << 4) + h;
    a[h] = (_Float16)src[r * ld + k];
  }
  return a;
}
// B 32x16 (KxN) from N-major storage (element (k,n) at src[n*ld + k]).
// lane = 16*g + r holds column n=r ; half h -> K = 16*g + h  (contiguous read!)
static __device__ __forceinline__ v16h load_bt_f16(const _Float16* src, int ld,
                                                   int g, int r) {
  v16h b;
#pragma unroll
  for (int h = 0; h < 16; ++h) b[h] = src[r * ld + (g << 4) + h];
  return b;
}
// B 32x16 (KxN) from K-major storage (element (k,n) at src[k*ld + n]).
static __device__ __forceinline__ v16h load_b_f16(const _Float16* src, int ld,
                                                  int g, int r) {
  v16h b;
#pragma unroll
  for (int h = 0; h < 16; ++h) b[h] = src[((g << 4) + h) * ld + r];
  return b;
}

// ---- fp32 -> f16 conversion ----
__global__ void cvt_f32_f16(const float* __restrict__ x, _Float16* __restrict__ y,
                            int n) {
  int i = blockIdx.x * blockDim.x + threadIdx.x;
  if (i < n) y[i] = (_Float16)x[i];
}

// ---- per-head input projection: Y[n][h][s][o] = sum_d X[n][s][h*64+d] * W[o][d]
__global__ void __launch_bounds__(128)
proj_kernel(const float* __restrict__ X, const _Float16* __restrict__ W16,
            _Float16* __restrict__ Y) {
  const int lane = threadIdx.x & 31;
  const int g = lane >> 4, r = lane & 15;
  const int wave = threadIdx.x >> 5;
  int task = blockIdx.x * 4 + wave;          // NB*HEADS*(SEQL/16) tasks
  const int mt = task & 63; task >>= 6;      // seq tile
  const int h = task & 15;
  const int n = task >> 4;

  const float* Xt = X + ((size_t)(n * SEQL + mt * 16)) * EMBED + h * HDIM;
  _Float16* Yt = Y + (((size_t)(n * HEADS + h)) * SEQL + mt * 16) * HDIM;

  v8f acc[4] = {};
#pragma unroll
  for (int kc = 0; kc < 2; ++kc) {
    v16h a = load_a_f32(Xt + kc * 32, EMBED, g, r);
#pragma unroll
    for (int nt = 0; nt < 4; ++nt) {
      v16h b = load_bt_f16(W16 + nt * 16 * HDIM + kc * 32, HDIM, g, r);
      acc[nt] = wmma_f16(a, b, acc[nt]);
    }
  }
#pragma unroll
  for (int nt = 0; nt < 4; ++nt)
#pragma unroll
    for (int j = 0; j < 8; ++j)
      Yt[(8 * g + j) * HDIM + nt * 16 + r] = (_Float16)acc[nt][j];
}

// ---- attention for one (n, h, 16-row q tile). 128 threads = 4 waves. ----
__global__ void __launch_bounds__(128)
attn_kernel(const _Float16* __restrict__ Qp, const _Float16* __restrict__ Kp,
            const _Float16* __restrict__ Vp, const int* __restrict__ mask,
            _Float16* __restrict__ AO) {
  __shared__ float    S[16][SEQL];  // 64 KB raw scores / exp values
  __shared__ _Float16 P[16][SEQL];  // 32 KB normalized probabilities (f16)

  const int lane = threadIdx.x & 31;
  const int g = lane >> 4, r = lane & 15;
  const int wave = threadIdx.x >> 5;
  int blk = blockIdx.x;
  const int qt = blk & 63; blk >>= 6;
  const int h = blk & 15;
  const int n = blk >> 4;

  const _Float16* Qh = Qp + (((size_t)(n * HEADS + h)) * SEQL + qt * 16) * HDIM;
  const _Float16* Kh = Kp + ((size_t)(n * HEADS + h)) * SEQL * HDIM;
  const _Float16* Vh = Vp + ((size_t)(n * HEADS + h)) * SEQL * HDIM;

  // Phase 1: S = Q * K^T, wave covers key columns [wave*256, wave*256+256)
  v16h qa0 = load_a_f16(Qh, HDIM, g, r);
  v16h qa1 = load_a_f16(Qh + 32, HDIM, g, r);
#pragma unroll 4
  for (int t = 0; t < 16; ++t) {
    const int n0 = wave * 256 + t * 16;
    __builtin_prefetch(Kh + (size_t)(n0 + 16) * HDIM, 0, 3);
    v8f acc = {};
    acc = wmma_f16(qa0, load_bt_f16(Kh + (size_t)n0 * HDIM, HDIM, g, r), acc);
    acc = wmma_f16(qa1, load_bt_f16(Kh + (size_t)n0 * HDIM + 32, HDIM, g, r), acc);
#pragma unroll
    for (int j = 0; j < 8; ++j) S[8 * g + j][n0 + r] = acc[j];
  }
  __syncthreads();

  // Phase 2: masked, scaled softmax. 8 consecutive lanes cooperate per row.
  const int row = threadIdx.x >> 3;
  const int t8 = threadIdx.x & 7;
  const float scale = 0.03125f;  // 1/sqrt(EMBED)
  const int* mrow = mask + n * SEQL;
  float m = -3.0e38f;
  for (int c = t8; c < SEQL; c += 8) {
    float s = (mrow[c] == 0) ? -1.0e20f : S[row][c] * scale;
    S[row][c] = s;
    m = fmaxf(m, s);
  }
  m = fmaxf(m, __shfl_xor(m, 1, 32));
  m = fmaxf(m, __shfl_xor(m, 2, 32));
  m = fmaxf(m, __shfl_xor(m, 4, 32));
  float sum = 0.0f;
  for (int c = t8; c < SEQL; c += 8) {
    float p = __expf(S[row][c] - m);
    S[row][c] = p;
    sum += p;
  }
  sum += __shfl_xor(sum, 1, 32);
  sum += __shfl_xor(sum, 2, 32);
  sum += __shfl_xor(sum, 4, 32);
  const float inv = 1.0f / sum;
  // normalize + convert to f16 once; PV loop below reads f16 only
  for (int c = t8; c < SEQL; c += 8) P[row][c] = (_Float16)(S[row][c] * inv);
  __syncthreads();

  // Phase 3: O = P * V, wave computes output columns [wave*16, wave*16+16)
  v8f acc = {};
  for (int kc = 0; kc < 32; ++kc) {
    __builtin_prefetch(Vh + (size_t)(kc + 1) * 32 * HDIM + wave * 16, 0, 3);
    v16h a = load_a_f16(&P[0][0] + kc * 32, SEQL, g, r);
    v16h b = load_b_f16(Vh + (size_t)kc * 32 * HDIM + wave * 16, HDIM, g, r);
    acc = wmma_f16(a, b, acc);
  }
  _Float16* Ao = AO + ((size_t)(n * SEQL + qt * 16)) * EMBED + h * HDIM + wave * 16;
#pragma unroll
  for (int j = 0; j < 8; ++j)
    Ao[(size_t)(8 * g + j) * EMBED + r] = (_Float16)acc[j];
}

// ---- output projection: Y = A(4096x1024 f16) @ W_out^T + b_out, fp32 out ----
__global__ void __launch_bounds__(128)
outproj_kernel(const _Float16* __restrict__ A, const _Float16* __restrict__ Wo16,
               const float* __restrict__ bias, float* __restrict__ Y) {
  const int lane = threadIdx.x & 31;
  const int g = lane >> 4, r = lane & 15;
  const int wave = threadIdx.x >> 5;
  int task = blockIdx.x * 4 + wave;      // (NB*SEQL/16) * (EMBED/64) tasks
  const int cg = task & 15;              // 64-column group
  const int mt = task >> 4;              // 16-row tile

  const _Float16* At = A + (size_t)mt * 16 * EMBED;
  v8f acc[4] = {};
  for (int kc = 0; kc < 32; ++kc) {
    v16h a = load_a_f16(At + kc * 32, EMBED, g, r);
#pragma unroll
    for (int nt = 0; nt < 4; ++nt) {
      v16h b = load_bt_f16(Wo16 + (size_t)(cg * 64 + nt * 16) * EMBED + kc * 32,
                           EMBED, g, r);
      acc[nt] = wmma_f16(a, b, acc[nt]);
    }
  }
#pragma unroll
  for (int nt = 0; nt < 4; ++nt) {
    const int col = cg * 64 + nt * 16 + r;
    const float bb = bias[col];
#pragma unroll
    for (int j = 0; j < 8; ++j)
      Y[(size_t)(mt * 16 + 8 * g + j) * EMBED + col] = acc[nt][j] + bb;
  }
}

extern "C" void kernel_launch(void* const* d_in, const int* in_sizes, int n_in,
                              void* d_out, int out_size, void* d_ws, size_t ws_size,
                              hipStream_t stream) {
  const float* values_x  = (const float*)d_in[0];
  const float* keys_x    = (const float*)d_in[1];
  const float* queries_x = (const float*)d_in[2];
  const int*   mask      = (const int*)d_in[3];
  const float* Wv    = (const float*)d_in[4];
  const float* Wk    = (const float*)d_in[5];
  const float* Wq    = (const float*)d_in[6];
  const float* W_out = (const float*)d_in[7];
  const float* b_out = (const float*)d_in[8];
  float* out = (float*)d_out;

  char* ws = (char*)d_ws;
  size_t off = 0;
  const size_t headTensor = (size_t)NB * HEADS * SEQL * HDIM * sizeof(_Float16); // 8 MB
  _Float16* Qp = (_Float16*)(ws + off); off += headTensor;
  _Float16* Kp = (_Float16*)(ws + off); off += headTensor;
  _Float16* Vp = (_Float16*)(ws + off); off += headTensor;
  _Float16* AO = (_Float16*)(ws + off); off += (size_t)NB * SEQL * EMBED * sizeof(_Float16);
  _Float16* Wq16 = (_Float16*)(ws + off); off += (size_t)HDIM * HDIM * sizeof(_Float16);
  _Float16* Wk16 = (_Float16*)(ws + off); off += (size_t)HDIM * HDIM * sizeof(_Float16);
  _Float16* Wv16 = (_Float16*)(ws + off); off += (size_t)HDIM * HDIM * sizeof(_Float16);
  _Float16* Wo16 = (_Float16*)(ws + off); off += (size_t)EMBED * EMBED * sizeof(_Float16);

  // 1) weight conversion to f16
  cvt_f32_f16<<<(HDIM * HDIM + 255) / 256, 256, 0, stream>>>(Wq, Wq16, HDIM * HDIM);
  cvt_f32_f16<<<(HDIM * HDIM + 255) / 256, 256, 0, stream>>>(Wk, Wk16, HDIM * HDIM);
  cvt_f32_f16<<<(HDIM * HDIM + 255) / 256, 256, 0, stream>>>(Wv, Wv16, HDIM * HDIM);
  cvt_f32_f16<<<(EMBED * EMBED + 255) / 256, 256, 0, stream>>>(W_out, Wo16, EMBED * EMBED);

  // 2) Q/K/V per-head projections (tasks = 4*16*64 = 4096 waves -> 1024 blocks)
  proj_kernel<<<1024, 128, 0, stream>>>(queries_x, Wq16, Qp);
  proj_kernel<<<1024, 128, 0, stream>>>(keys_x,    Wk16, Kp);
  proj_kernel<<<1024, 128, 0, stream>>>(values_x,  Wv16, Vp);

  // 3) attention: one block per (n, h, q-tile) = 4*16*64 = 4096 blocks
  attn_kernel<<<4096, 128, 0, stream>>>(Qp, Kp, Vp, mask, AO);

  // 4) output projection (tasks = 256*16 = 4096 waves -> 1024 blocks)
  outproj_kernel<<<1024, 128, 0, stream>>>(AO, Wo16, b_out, out);
}
